// MHA_16045997818164
// MI455X (gfx1250) — matrix operand order repaired and checked by reading, compiled, not run on hardware
//
#include <hip/hip_runtime.h>
#include <hip/hip_bf16.h>

typedef __attribute__((ext_vector_type(16))) __bf16 v16bf;
typedef __attribute__((ext_vector_type(8)))  float  v8f;

#define HID   1024
#define NH    16
#define HD    64
#define SEQ   2048
#define BATCH 2
#define NTOK  (BATCH * SEQ)

#define WMMA_BF16(A, B, C) \
  __builtin_amdgcn_wmma_f32_16x16x32_bf16(false, (A), false, (B), (short)0, (C), false, false)

union Frag {
  v16bf v;
  uint4 q[2];
  unsigned int d[8];
};

static __device__ __forceinline__ v8f vzero8() {
  v8f z = {0.f, 0.f, 0.f, 0.f, 0.f, 0.f, 0.f, 0.f};
  return z;
}

static __device__ __forceinline__ unsigned int pack2(float a, float b) {
  union { __bf16 h[2]; unsigned int u; } p;
  p.h[0] = (__bf16)a;
  p.h[1] = (__bf16)b;
  return p.u;
}

// ---------------------------------------------------------------------------
// fp32 -> bf16 conversion, 4 elements per thread (n must be a multiple of 4)
// ---------------------------------------------------------------------------
__global__ void cvt_f32_bf16_x4(const float* __restrict__ in,
                                __bf16* __restrict__ out, int n4) {
  int i = blockIdx.x * blockDim.x + threadIdx.x;
  if (i >= n4) return;
  const float4 f = reinterpret_cast<const float4*>(in)[i];
  union { __bf16 h[4]; uint2 u; } p;
  p.h[0] = (__bf16)f.x; p.h[1] = (__bf16)f.y;
  p.h[2] = (__bf16)f.z; p.h[3] = (__bf16)f.w;
  reinterpret_cast<uint2*>(out)[i] = p.u;
}

// ---------------------------------------------------------------------------
// QKV GEMM (M=4096, N=3072, K=1024) + bias + RoPE + pack.
// One wave computes a 32(token) x 64(col) tile == one (which, head) d-slice
// for two 16-token groups; B-fragments amortized over both.
// Q/K packed [B,H,S,D] bf16; V packed transposed [B,H,D,S] bf16.
// ---------------------------------------------------------------------------
__global__ void qkv_rope_kernel(const __bf16* __restrict__ xb,
                                const __bf16* __restrict__ wb,
                                const float*  __restrict__ bias,
                                __bf16* __restrict__ qo,
                                __bf16* __restrict__ ko,
                                __bf16* __restrict__ vto) {
  const int wave = (blockIdx.x * blockDim.x + threadIdx.x) >> 5;
  const int lane = threadIdx.x & 31;
  const int hw   = lane >> 4;   // half-wave: 0 or 1
  const int l15  = lane & 15;
  const int nblock = wave % 48; // 48 blocks of 64 output cols
  const int mtile  = wave / 48; // 128 tiles of 32 token rows

  v8f acc[2][4];
#pragma unroll
  for (int mi = 0; mi < 2; ++mi)
#pragma unroll
    for (int t = 0; t < 4; ++t) acc[mi][t] = vzero8();

  // A: x rows; contiguous k order (16*hw + s) -> one 32B chunk at k + 16*hw
  const __bf16* arow[2];
#pragma unroll
  for (int mi = 0; mi < 2; ++mi)
    arow[mi] = xb + (size_t)(mtile * 32 + mi * 16 + l15) * HID;
  // B: W^T -> W rows; same contiguous k order as A -> one 32B chunk at k + 16*hw
  const __bf16* brow[4];
#pragma unroll
  for (int t = 0; t < 4; ++t)
    brow[t] = wb + (size_t)(nblock * 64 + t * 16 + l15) * HID + 16 * hw;

  for (int kb = 0; kb < HID; kb += 32) {
    Frag a[2];
#pragma unroll
    for (int mi = 0; mi < 2; ++mi) {
      a[mi].q[0] = *reinterpret_cast<const uint4*>(arow[mi] + kb + 16 * hw);
      a[mi].q[1] = *reinterpret_cast<const uint4*>(arow[mi] + kb + 16 * hw + 8);
      __builtin_prefetch(arow[mi] + kb + 128, 0, 0);
    }
#pragma unroll
    for (int t = 0; t < 4; ++t) {
      Frag b;
      b.q[0] = *reinterpret_cast<const uint4*>(brow[t] + kb);
      b.q[1] = *reinterpret_cast<const uint4*>(brow[t] + kb + 8);
      acc[0][t] = WMMA_BF16(a[0].v, b.v, acc[0][t]);
      acc[1][t] = WMMA_BF16(a[1].v, b.v, acc[1][t]);
    }
  }

  // Epilogue: C layout (32-bit 16x16): element r, lane l -> row = r + 8*hw,
  // col = l15 (within tile t).
  const int ncolbase = nblock * 64;
  const int which = ncolbase >> 10;      // 0=q 1=k 2=v
  const int h = (ncolbase >> 6) & 15;    // head

#pragma unroll
  for (int mi = 0; mi < 2; ++mi) {
    float vals[4][8];
#pragma unroll
    for (int t = 0; t < 4; ++t) {
      const float bia = bias[ncolbase + t * 16 + l15];
#pragma unroll
      for (int r = 0; r < 8; ++r) vals[t][r] = acc[mi][t][r] + bia;
    }

    if (which == 2) {
      // V: store transposed [B,H,D,S]
#pragma unroll
      for (int t = 0; t < 4; ++t) {
        const int d = t * 16 + l15;
#pragma unroll
        for (int r = 0; r < 8; ++r) {
          const int token = mtile * 32 + mi * 16 + r + 8 * hw;
          const int b = token >> 11, s = token & (SEQ - 1);
          vto[((size_t)(b * NH + h) * HD + d) * SEQ + s] = (__bf16)vals[t][r];
        }
      }
    } else {
      __bf16* dst = (which == 0) ? qo : ko;
      // rotate_half: d<32 -> -x[d+32] (tile t+2); d>=32 -> x[d-32] (tile t-2)
#pragma unroll
      for (int t = 0; t < 4; ++t) {
        const int d = t * 16 + l15;
        const int i = d & 31;
        const float invf = __expf(-0.28782313662425573f * (float)i); // 10000^(-i/32)
#pragma unroll
        for (int r = 0; r < 8; ++r) {
          const int token = mtile * 32 + mi * 16 + r + 8 * hw;
          const int b = token >> 11, s = token & (SEQ - 1);
          const float ang = (float)s * invf;
          const float c = __cosf(ang), sn = __sinf(ang);
          const float rot = (t < 2) ? -vals[t + 2][r] : vals[t - 2][r];
          dst[((size_t)(b * NH + h) * SEQ + s) * HD + d] =
              (__bf16)(vals[t][r] * c + rot * sn);
        }
      }
    }
  }
}

// ---------------------------------------------------------------------------
// Flash attention: one wave per (b, h, 32-query tile): two 16-query groups
// share every K and V^T fragment load.
// scores^T = K_tile(16k x 64d) x Q^T(64d x 16q)  -> per-query stats per lane.
// PV: acc(16d x 16q) += V^T(16d x 32k) x P^T(32k x 16q).
// ---------------------------------------------------------------------------
__global__ void attn_kernel(const __bf16* __restrict__ qg,
                            const __bf16* __restrict__ kg,
                            const __bf16* __restrict__ vtg,
                            __bf16* __restrict__ ao) {
  const int wave = (blockIdx.x * blockDim.x + threadIdx.x) >> 5;
  const int lane = threadIdx.x & 31;
  const int hw   = lane >> 4;
  const int l15  = lane & 15;
  const int qtile = wave & 63;  // 64 tiles of 32 queries
  const int bh    = wave >> 6;  // b*16 + h

  const __bf16* qb = qg  + (size_t)bh * SEQ * HD;
  const __bf16* kp = kg  + (size_t)bh * SEQ * HD;
  const __bf16* vb = vtg + (size_t)bh * HD * SEQ;

  // Q^T B-fragments per q-group (d-chunks 0..31, 32..63): contiguous 32B/lane.
  Frag bq[2][2];
#pragma unroll
  for (int g = 0; g < 2; ++g) {
    const __bf16* qr = qb + (size_t)(qtile * 32 + g * 16 + l15) * HD;
    bq[g][0].q[0] = *reinterpret_cast<const uint4*>(qr + 16 * hw);
    bq[g][0].q[1] = *reinterpret_cast<const uint4*>(qr + 16 * hw + 8);
    bq[g][1].q[0] = *reinterpret_cast<const uint4*>(qr + 32 + 16 * hw);
    bq[g][1].q[1] = *reinterpret_cast<const uint4*>(qr + 32 + 16 * hw + 8);
  }

  v8f acc[2][4];
#pragma unroll
  for (int g = 0; g < 2; ++g)
#pragma unroll
    for (int t = 0; t < 4; ++t) acc[g][t] = vzero8();
  float m_run[2] = {-3.0e38f, -3.0e38f};
  float l_run[2] = {0.f, 0.f};
  const float SCALE = 0.125f; // 1/sqrt(64)

  for (int kb = 0; kb < SEQ; kb += 32) {
    // ---- scores^T: two 16-key tiles x two q-groups; K frags shared ----
    v8f s[2][2]; // [q-group][key-tile]
    s[0][0] = vzero8(); s[0][1] = vzero8();
    s[1][0] = vzero8(); s[1][1] = vzero8();
#pragma unroll
    for (int j = 0; j < 2; ++j) {
      const __bf16* kr = kp + (size_t)(kb + j * 16 + l15) * HD;
      Frag a;
      a.q[0] = *reinterpret_cast<const uint4*>(kr + 16 * hw);
      a.q[1] = *reinterpret_cast<const uint4*>(kr + 16 * hw + 8);
      s[0][j] = WMMA_BF16(a.v, bq[0][0].v, s[0][j]);
      s[1][j] = WMMA_BF16(a.v, bq[1][0].v, s[1][j]);
      a.q[0] = *reinterpret_cast<const uint4*>(kr + 32 + 16 * hw);
      a.q[1] = *reinterpret_cast<const uint4*>(kr + 40 + 16 * hw);
      s[0][j] = WMMA_BF16(a.v, bq[0][1].v, s[0][j]);
      s[1][j] = WMMA_BF16(a.v, bq[1][1].v, s[1][j]);
    }

    // ---- online softmax per q-group (per query = lane pair (l, l^16)) ----
    Frag bp[2];
    float corr[2];
#pragma unroll
    for (int g = 0; g < 2; ++g) {
      float mloc = -3.0e38f;
#pragma unroll
      for (int r = 0; r < 8; ++r) {
        s[g][0][r] *= SCALE; s[g][1][r] *= SCALE;
        mloc = fmaxf(mloc, fmaxf(s[g][0][r], s[g][1][r]));
      }
      mloc = fmaxf(mloc, __shfl_xor(mloc, 16, 32));
      const float m_new = fmaxf(m_run[g], mloc);
      corr[g] = __expf(m_run[g] - m_new);
      m_run[g] = m_new;
      l_run[g] *= corr[g];

      unsigned int p0[4], p1[4];
#pragma unroll
      for (int v2 = 0; v2 < 4; ++v2) {
        const float e0 = __expf(s[g][0][2 * v2] - m_new);
        const float e1 = __expf(s[g][0][2 * v2 + 1] - m_new);
        const float f0 = __expf(s[g][1][2 * v2] - m_new);
        const float f1 = __expf(s[g][1][2 * v2 + 1] - m_new);
        l_run[g] += e0 + e1 + f0 + f1;
        p0[v2] = pack2(e0, e1);
        p1[v2] = pack2(f0, f1);
      }

      // build P^T B-fragment (32k x 16q) via lane-pair exchange:
      // lanes<16 need keys kb+0..15 : own s0 (0..7) + partner s0 (8..15)
      // lanes>=16 need keys kb+16..31: partner s1 (16..23) + own s1 (24..31)
#pragma unroll
      for (int v2 = 0; v2 < 4; ++v2) {
        const unsigned int x0 = (unsigned int)__shfl_xor((int)p0[v2], 16, 32);
        const unsigned int x1 = (unsigned int)__shfl_xor((int)p1[v2], 16, 32);
        bp[g].d[v2]     = hw ? x1 : p0[v2];
        bp[g].d[v2 + 4] = hw ? p1[v2] : x0;
      }
    }

#pragma unroll
    for (int g = 0; g < 2; ++g)
#pragma unroll
      for (int t = 0; t < 4; ++t)
#pragma unroll
        for (int r = 0; r < 8; ++r) acc[g][t][r] *= corr[g];

    // ---- PV: V^T A-fragments (rows of [B,H,D,S]) shared by both groups ----
#pragma unroll
    for (int t = 0; t < 4; ++t) {
      const __bf16* vr = vb + (size_t)(t * 16 + l15) * SEQ + kb;
      Frag av;
      av.q[0] = *reinterpret_cast<const uint4*>(vr + 16 * hw);
      av.q[1] = *reinterpret_cast<const uint4*>(vr + 16 * hw + 8);
      acc[0][t] = WMMA_BF16(av.v, bp[0].v, acc[0][t]);
      acc[1][t] = WMMA_BF16(av.v, bp[1].v, acc[1][t]);
    }
  }

  const int b = bh >> 4, h = bh & 15;
#pragma unroll
  for (int g = 0; g < 2; ++g) {
    const float l_tot = l_run[g] + __shfl_xor(l_run[g], 16, 32);
    const float inv = 1.f / l_tot;
    const int token = b * SEQ + qtile * 32 + g * 16 + l15;
#pragma unroll
    for (int t = 0; t < 4; ++t)
#pragma unroll
      for (int r = 0; r < 8; ++r) {
        const int d = t * 16 + r + 8 * hw;
        ao[(size_t)token * HID + h * HD + d] = (__bf16)(acc[g][t][r] * inv);
      }
  }
}

// ---------------------------------------------------------------------------
// Output projection: out[4096,1024] = attn_bf16 @ out_w^T + out_b (fp32 out)
// One wave computes 32(token) x 64(col).
// ---------------------------------------------------------------------------
__global__ void out_gemm_kernel(const __bf16* __restrict__ ab,
                                const __bf16* __restrict__ wb,
                                const float*  __restrict__ bias,
                                float* __restrict__ out) {
  const int wave = (blockIdx.x * blockDim.x + threadIdx.x) >> 5;
  const int lane = threadIdx.x & 31;
  const int hw   = lane >> 4;
  const int l15  = lane & 15;
  const int nblock = wave % 16; // 16 blocks of 64 cols
  const int mtile  = wave / 16; // 128 tiles of 32 rows

  v8f acc[2][4];
#pragma unroll
  for (int mi = 0; mi < 2; ++mi)
#pragma unroll
    for (int t = 0; t < 4; ++t) acc[mi][t] = vzero8();

  const __bf16* arow[2];
#pragma unroll
  for (int mi = 0; mi < 2; ++mi)
    arow[mi] = ab + (size_t)(mtile * 32 + mi * 16 + l15) * HID;
  const __bf16* brow[4];
#pragma unroll
  for (int t = 0; t < 4; ++t)
    brow[t] = wb + (size_t)(nblock * 64 + t * 16 + l15) * HID + 16 * hw;

  for (int kb = 0; kb < HID; kb += 32) {
    Frag a[2];
#pragma unroll
    for (int mi = 0; mi < 2; ++mi) {
      a[mi].q[0] = *reinterpret_cast<const uint4*>(arow[mi] + kb + 16 * hw);
      a[mi].q[1] = *reinterpret_cast<const uint4*>(arow[mi] + kb + 16 * hw + 8);
      __builtin_prefetch(arow[mi] + kb + 128, 0, 0);
    }
#pragma unroll
    for (int t = 0; t < 4; ++t) {
      Frag b;
      b.q[0] = *reinterpret_cast<const uint4*>(brow[t] + kb);
      b.q[1] = *reinterpret_cast<const uint4*>(brow[t] + kb + 8);
      acc[0][t] = WMMA_BF16(a[0].v, b.v, acc[0][t]);
      acc[1][t] = WMMA_BF16(a[1].v, b.v, acc[1][t]);
    }
  }

#pragma unroll
  for (int mi = 0; mi < 2; ++mi)
#pragma unroll
    for (int t = 0; t < 4; ++t) {
      const int col = nblock * 64 + t * 16 + l15;
      const float bia = bias[col];
#pragma unroll
      for (int r = 0; r < 8; ++r) {
        const int token = mtile * 32 + mi * 16 + r + 8 * hw;
        out[(size_t)token * HID + col] = acc[mi][t][r] + bia;
      }
    }
}

// ---------------------------------------------------------------------------
extern "C" void kernel_launch(void* const* d_in, const int* in_sizes, int n_in,
                              void* d_out, int out_size, void* d_ws, size_t ws_size,
                              hipStream_t stream) {
  const float* x      = (const float*)d_in[0];
  const float* Wqkv_w = (const float*)d_in[1];
  const float* Wqkv_b = (const float*)d_in[2];
  const float* out_w  = (const float*)d_in[3];
  const float* out_b  = (const float*)d_in[4];

  char* ws = (char*)d_ws;
  __bf16* xb    = (__bf16*)(ws +         0); //  8 MiB: x bf16 [4096,1024]
  __bf16* wqkvb = (__bf16*)(ws +  8388608); //  6 MiB: Wqkv bf16 [3072,1024]
  __bf16* outwb = (__bf16*)(ws + 14680064); //  2 MiB: out_w bf16 [1024,1024]
  __bf16* qws   = (__bf16*)(ws + 16777216); //  8 MiB: Q bf16 [B,H,S,D]
  __bf16* kws   = (__bf16*)(ws + 25165824); //  8 MiB: K bf16 [B,H,S,D]
  __bf16* vtws  = (__bf16*)(ws + 33554432); //  8 MiB: V^T bf16 [B,H,D,S]
  __bf16* attnb = (__bf16*)(ws + 41943040); //  8 MiB: attn out bf16 [4096,1024]

  const int nx = NTOK * HID / 4;      // 1048576 float4s
  const int nw = 3 * HID * HID / 4;   //  786432
  const int no = HID * HID / 4;       //  262144
  cvt_f32_bf16_x4<<<(nx + 255) / 256, 256, 0, stream>>>(x, xb, nx);
  cvt_f32_bf16_x4<<<(nw + 255) / 256, 256, 0, stream>>>(Wqkv_w, wqkvb, nw);
  cvt_f32_bf16_x4<<<(no + 255) / 256, 256, 0, stream>>>(out_w, outwb, no);

  // 128 m-tiles x 48 n-blocks = 6144 waves, 8 waves/block
  qkv_rope_kernel<<<768, 256, 0, stream>>>(xb, wqkvb, Wqkv_b, qws, kws, vtws);
  // 2*16*64 = 2048 waves
  attn_kernel<<<256, 256, 0, stream>>>(qws, kws, vtws, attnb);
  // 128 m-tiles x 16 n-blocks = 2048 waves
  out_gemm_kernel<<<256, 256, 0, stream>>>(attnb, outwb, out_b, (float*)d_out);
}
